// Gemma3nTextAttention_40346922778916
// MI455X (gfx1250) — matrix operand order repaired and checked by reading, not compiled, not run on hardware
//
#include <hip/hip_runtime.h>
#include <cstdint>
#include <cstddef>

// ---------------- problem constants (Gemma3n attention) ----------------
#define B_    2
#define S_    2048
#define HID_  2048
#define NH_   8
#define NKV_  2
#define HD_   256
#define NREP_ (NH_ / NKV_)
#define BS_   (B_ * S_)

// ---------------- WMMA types ----------------
typedef __attribute__((ext_vector_type(16))) __bf16 v16bf;
typedef __attribute__((ext_vector_type(8)))  __bf16 v8bf;
typedef __attribute__((ext_vector_type(8)))  float  v8f;

__device__ __forceinline__ v8f zero8() {
  v8f z;
#pragma unroll
  for (int i = 0; i < 8; ++i) z[i] = 0.f;
  return z;
}

__device__ __forceinline__ __bf16 f2bf(float f) {
  union { float f; unsigned int u; } in;
  in.f = f;
  unsigned int u = in.u;
  unsigned int r = (u + 0x7FFFu + ((u >> 16) & 1u)) >> 16;  // RNE
  union { unsigned short u; __bf16 b; } out;
  out.u = (unsigned short)r;
  return out.b;
}

__device__ __forceinline__ v8f wmma_bf16(v16bf a, v16bf b, v8f c) {
  // D(f32 16x16) = A(16x32 bf16) x B(32x16 bf16) + C
  return __builtin_amdgcn_wmma_f32_16x16x32_bf16(false, a, false, b,
                                                 (short)0, c, false, false);
}

// A fragment 16x32 (MxK): lane L -> row (L&15); half hf=L>=16;
// elements 0..15 = K[16*hf + 0..15]
__device__ __forceinline__ v16bf load_fragA(const __bf16* base, size_t stride) {
  const int lane = (int)(threadIdx.x & 31u);
  const __bf16* p = base + (size_t)(lane & 15) * stride + (size_t)((lane >> 4) << 4);
  v8bf lo = *(const v8bf*)(p);
  v8bf hi = *(const v8bf*)(p + 8);
  return __builtin_shufflevector(lo, hi, 0, 1, 2, 3, 4, 5, 6, 7,
                                 8, 9, 10, 11, 12, 13, 14, 15);
}

// B fragment 32x16 (KxN) gathered from an N-major source (row n holds contiguous K):
// lane L -> col (L&15); elements 0..15 = K[16*hf + 0..15]
__device__ __forceinline__ v16bf load_fragB(const __bf16* base, size_t stride) {
  const int lane = (int)(threadIdx.x & 31u);
  const __bf16* p = base + (size_t)(lane & 15) * stride + (size_t)((lane >> 4) << 4);
  v8bf lo = *(const v8bf*)(p);
  v8bf hi = *(const v8bf*)(p + 8);
  return __builtin_shufflevector(lo, hi, 0, 1, 2, 3, 4, 5, 6, 7,
                                 8, 9, 10, 11, 12, 13, 14, 15);
}

// ---------------- f32 -> bf16 conversion ----------------
__global__ __launch_bounds__(256) void f32_to_bf16_kernel(const float* __restrict__ in,
                                                          __bf16* __restrict__ out, int n) {
  int i = blockIdx.x * blockDim.x + threadIdx.x;
  int stride = gridDim.x * blockDim.x;
  for (; i < n; i += stride) out[i] = f2bf(in[i]);
}

// ---------------- NT GEMM: C[M][N] = A[M][K] * W[N][K]^T (bf16 in, f32 out) --------
// 128 threads = 4 waves (2x2); block tile 128x128; wave tile 64x64 (4x4 C tiles)
// Per k-step/wave: 4 A-frags + 4 B-frags (16 b128 loads/lane) feed 16 WMMAs
// -> ~32 flops per L2 byte; operands fully L2-resident (192 MB L2).
__global__ __launch_bounds__(128) void gemm_nt_bf16_f32(
    const __bf16* __restrict__ A, const __bf16* __restrict__ W,
    float* __restrict__ C, int M, int N, int K) {
  const int lane = (int)(threadIdx.x & 31u);
  const int wave = (int)(threadIdx.x >> 5);
  const int m0 = blockIdx.y * 128 + (wave & 1) * 64;
  const int n0 = blockIdx.x * 128 + (wave >> 1) * 64;

  v8f acc[4][4];
#pragma unroll
  for (int i = 0; i < 4; ++i)
#pragma unroll
    for (int j = 0; j < 4; ++j) acc[i][j] = zero8();

  for (int kc = 0; kc < K; kc += 32) {
    v16bf a[4];
    v16bf b[4];
#pragma unroll
    for (int mt = 0; mt < 4; ++mt)
      a[mt] = load_fragA(A + (size_t)(m0 + mt * 16) * K + kc, (size_t)K);
#pragma unroll
    for (int nt = 0; nt < 4; ++nt)
      b[nt] = load_fragB(W + (size_t)(n0 + nt * 16) * K + kc, (size_t)K);
#pragma unroll
    for (int mt = 0; mt < 4; ++mt)
#pragma unroll
      for (int nt = 0; nt < 4; ++nt)
        acc[mt][nt] = wmma_bf16(a[mt], b[nt], acc[mt][nt]);
  }

  const int hf = lane >> 4;
  const int nn = lane & 15;
#pragma unroll
  for (int mt = 0; mt < 4; ++mt)
#pragma unroll
    for (int nt = 0; nt < 4; ++nt)
#pragma unroll
      for (int r = 0; r < 8; ++r) {
        int mi = m0 + mt * 16 + r + hf * 8;
        int ni = n0 + nt * 16 + nn;
        C[(size_t)mi * N + ni] = acc[mt][nt][r];
      }
}

// ---------------- LayerNorm + RoPE ----------------
// one block per (b,s); 4 waves x 3 heads = 12 heads (8 q, 2 k, 2 v)
// lane owns d = lane*8 .. lane*8+7; RoPE partner (d xor 128) lives at lane xor 16
__global__ __launch_bounds__(128) void ln_rope_kernel(
    const float* __restrict__ qf, const float* __restrict__ kf,
    const float* __restrict__ vf, const float* __restrict__ cosb,
    const float* __restrict__ sinb, const float* __restrict__ qw,
    const float* __restrict__ kw, __bf16* __restrict__ qT,
    __bf16* __restrict__ kT, __bf16* __restrict__ vT) {
  const int bs = blockIdx.x;
  const int b = bs / S_;
  const int s = bs - b * S_;
  const int lane = (int)(threadIdx.x & 31u);
  const int wave = (int)(threadIdx.x >> 5);
  const int dbase = lane * 8;

#pragma unroll
  for (int i = 0; i < 3; ++i) {
    const int hh = wave * 3 + i;  // 0..7 q, 8..9 k, 10..11 v
    const float* src;
    const float* w;
    if (hh < 8) {
      src = qf + (size_t)bs * (NH_ * HD_) + hh * HD_;
      w = qw;
    } else if (hh < 10) {
      src = kf + (size_t)bs * (NKV_ * HD_) + (hh - 8) * HD_;
      w = kw;
    } else {
      src = vf + (size_t)bs * (NKV_ * HD_) + (hh - 10) * HD_;
      w = nullptr;
    }

    float x[8];
#pragma unroll
    for (int j = 0; j < 8; ++j) x[j] = src[dbase + j];

    float sum = 0.f;
#pragma unroll
    for (int j = 0; j < 8; ++j) sum += x[j];
#pragma unroll
    for (int off = 16; off >= 1; off >>= 1) sum += __shfl_xor(sum, off, 32);
    const float mean = sum * (1.f / 256.f);

    float vs = 0.f;
#pragma unroll
    for (int j = 0; j < 8; ++j) {
      x[j] -= mean;
      vs += x[j] * x[j];
    }
#pragma unroll
    for (int off = 16; off >= 1; off >>= 1) vs += __shfl_xor(vs, off, 32);
    const float inv = rsqrtf(vs * (1.f / 256.f) + 1e-6f);

#pragma unroll
    for (int j = 0; j < 8; ++j) {
      float g = w ? w[dbase + j] : 1.f;
      x[j] = x[j] * inv * g;
    }

    if (hh < 10) {
      const float* cp = cosb + (size_t)bs * HD_;
      const float* sp = sinb + (size_t)bs * HD_;
#pragma unroll
      for (int j = 0; j < 8; ++j) {
        float other = __shfl_xor(x[j], 16, 32);       // value at d xor 128
        float rot = (lane < 16) ? -other : other;     // rot = [-x2, x1]
        x[j] = x[j] * cp[dbase + j] + rot * sp[dbase + j];
      }
      __bf16* dst =
          (hh < 8) ? qT + ((size_t)(b * NH_ + hh) * S_ + s) * HD_
                   : kT + ((size_t)(b * NKV_ + (hh - 8)) * S_ + s) * HD_;
#pragma unroll
      for (int j = 0; j < 8; ++j) dst[dbase + j] = f2bf(x[j]);
    } else {
      // V stored transposed: [b][kv][d][s]
      __bf16* dst = vT + (size_t)(b * NKV_ + (hh - 10)) * HD_ * S_;
#pragma unroll
      for (int j = 0; j < 8; ++j) dst[(size_t)(dbase + j) * S_ + s] = f2bf(x[j]);
    }
  }
}

// ---------------- flash attention (causal, no scale, GQA) ----------------
// one wave per 16-query tile; 32-key blocks; online softmax
__global__ __launch_bounds__(128) void attn_kernel(
    const __bf16* __restrict__ qT, const __bf16* __restrict__ kT,
    const __bf16* __restrict__ vT, __bf16* __restrict__ ctx) {
  __shared__ __bf16 pstage[4][16 * 32];  // per-wave P staging (C-layout -> A-layout)
  const int lane = (int)(threadIdx.x & 31u);
  const int wave = (int)(threadIdx.x >> 5);
  const int gw = blockIdx.x * 4 + wave;
  const int QT = S_ / 16;
  const int qt = gw % QT;
  const int bh = gw / QT;
  const int h = bh % NH_;
  const int b = bh / NH_;
  const int kv = h / NREP_;
  const int q0 = qt * 16;

  const __bf16* qhead = qT + ((size_t)(b * NH_ + h) * S_ + q0) * HD_;
  const __bf16* khead = kT + (size_t)(b * NKV_ + kv) * S_ * HD_;
  const __bf16* vhead = vT + (size_t)(b * NKV_ + kv) * HD_ * S_;

  v16bf qfrag[8];
#pragma unroll
  for (int kc = 0; kc < 8; ++kc) qfrag[kc] = load_fragA(qhead + kc * 32, HD_);

  v8f o[16];
#pragma unroll
  for (int t = 0; t < 16; ++t) o[t] = zero8();
  float mrow[8], lrow[8];
#pragma unroll
  for (int r = 0; r < 8; ++r) {
    mrow[r] = -1e30f;
    lrow[r] = 0.f;
  }

  const int hf = lane >> 4;
  const int nn = lane & 15;
  const int nkb = (q0 + 47) / 32;  // causal: keys 0 .. q0+15 only

  for (int kb = 0; kb < nkb; ++kb) {
    const int k0 = kb * 32;
    v8f s0 = zero8(), s1 = zero8();
#pragma unroll
    for (int kc = 0; kc < 8; ++kc)
      s0 = wmma_bf16(qfrag[kc],
                     load_fragB(khead + (size_t)k0 * HD_ + kc * 32, HD_), s0);
#pragma unroll
    for (int kc = 0; kc < 8; ++kc)
      s1 = wmma_bf16(qfrag[kc],
                     load_fragB(khead + (size_t)(k0 + 16) * HD_ + kc * 32, HD_), s1);

    if (k0 + 31 > q0) {  // diagonal block(s): apply causal mask (matches -1e9 mask)
#pragma unroll
      for (int r = 0; r < 8; ++r) {
        int qr = q0 + r + hf * 8;
        if (k0 + nn > qr) s0[r] += -1e9f;
        if (k0 + 16 + nn > qr) s1[r] += -1e9f;
      }
    }

    float scale[8];
#pragma unroll
    for (int r = 0; r < 8; ++r) {
      float mx = fmaxf(s0[r], s1[r]);
      mx = fmaxf(mx, __shfl_xor(mx, 1, 32));
      mx = fmaxf(mx, __shfl_xor(mx, 2, 32));
      mx = fmaxf(mx, __shfl_xor(mx, 4, 32));
      mx = fmaxf(mx, __shfl_xor(mx, 8, 32));  // row max within 16-lane half
      float mnew = fmaxf(mrow[r], mx);
      scale[r] = __expf(mrow[r] - mnew);
      mrow[r] = mnew;
      float p0 = __expf(s0[r] - mnew);
      float p1 = __expf(s1[r] - mnew);
      s0[r] = p0;
      s1[r] = p1;
      float rs = p0 + p1;
      rs += __shfl_xor(rs, 1, 32);
      rs += __shfl_xor(rs, 2, 32);
      rs += __shfl_xor(rs, 4, 32);
      rs += __shfl_xor(rs, 8, 32);
      lrow[r] = lrow[r] * scale[r] + rs;
    }
#pragma unroll
    for (int t = 0; t < 16; ++t)
#pragma unroll
      for (int r = 0; r < 8; ++r) o[t][r] *= scale[r];

    // stage P tile (16 queries x 32 keys) to LDS, reload in A-layout
    __bf16* pl = pstage[wave];
#pragma unroll
    for (int r = 0; r < 8; ++r) {
      pl[(hf * 8 + r) * 32 + nn] = f2bf(s0[r]);
      pl[(hf * 8 + r) * 32 + nn + 16] = f2bf(s1[r]);
    }
    v16bf pfrag = load_fragA(pl, 32);
#pragma unroll
    for (int dt = 0; dt < 16; ++dt)
      o[dt] = wmma_bf16(pfrag,
                        load_fragB(vhead + (size_t)(dt * 16) * S_ + k0, S_), o[dt]);
  }

#pragma unroll
  for (int r = 0; r < 8; ++r) lrow[r] = 1.f / lrow[r];
#pragma unroll
  for (int dt = 0; dt < 16; ++dt)
#pragma unroll
    for (int r = 0; r < 8; ++r) {
      int mi = q0 + r + hf * 8;
      ctx[((size_t)b * S_ + mi) * (NH_ * HD_) + h * HD_ + dt * 16 + nn] =
          f2bf(o[dt][r] * lrow[r]);
    }
}

// ---------------- host launcher ----------------
extern "C" void kernel_launch(void* const* d_in, const int* in_sizes, int n_in,
                              void* d_out, int out_size, void* d_ws, size_t ws_size,
                              hipStream_t stream) {
  (void)in_sizes; (void)n_in; (void)out_size; (void)ws_size;
  const float* hs   = (const float*)d_in[0];
  const float* cosb = (const float*)d_in[1];
  const float* sinb = (const float*)d_in[2];
  // d_in[3] attention_mask: pure causal -1e9 mask, computed analytically in-kernel
  const float* q_w  = (const float*)d_in[4];
  const float* k_w  = (const float*)d_in[5];
  const float* v_w  = (const float*)d_in[6];
  const float* o_w  = (const float*)d_in[7];
  const float* q_nw = (const float*)d_in[8];
  const float* k_nw = (const float*)d_in[9];
  float* out = (float*)d_out;

  char* ws = (char*)d_ws;
  size_t off = 0;
  auto alloc = [&](size_t bytes) -> char* {
    char* p = ws + off;
    off += (bytes + 255) & ~(size_t)255;
    return p;
  };

  __bf16* hsb = (__bf16*)alloc((size_t)BS_ * HID_ * 2);            // 16 MB
  __bf16* qwb = (__bf16*)alloc((size_t)NH_ * HD_ * HID_ * 2);      //  8 MB
  __bf16* kwb = (__bf16*)alloc((size_t)NKV_ * HD_ * HID_ * 2);     //  2 MB
  __bf16* vwb = (__bf16*)alloc((size_t)NKV_ * HD_ * HID_ * 2);     //  2 MB
  __bf16* owb = (__bf16*)alloc((size_t)HID_ * NH_ * HD_ * 2);      //  8 MB
  float*  qf  = (float*)alloc((size_t)BS_ * NH_ * HD_ * 4);        // 32 MB
  float*  kf  = (float*)alloc((size_t)BS_ * NKV_ * HD_ * 4);       //  8 MB
  float*  vf  = (float*)alloc((size_t)BS_ * NKV_ * HD_ * 4);       //  8 MB
  __bf16* qTb = (__bf16*)alloc((size_t)B_ * NH_ * S_ * HD_ * 2);   // 16 MB
  __bf16* kTb = (__bf16*)alloc((size_t)B_ * NKV_ * S_ * HD_ * 2);  //  4 MB
  __bf16* vTb = (__bf16*)alloc((size_t)B_ * NKV_ * S_ * HD_ * 2);  //  4 MB
  __bf16* ctx = (__bf16*)qf;  // reuse qf region (dead after ln_rope); 16 MB needed

  // 1) convert activations + weights to bf16
  f32_to_bf16_kernel<<<2048, 256, 0, stream>>>(hs, hsb, BS_ * HID_);
  f32_to_bf16_kernel<<<1024, 256, 0, stream>>>(q_w, qwb, NH_ * HD_ * HID_);
  f32_to_bf16_kernel<<<512, 256, 0, stream>>>(k_w, kwb, NKV_ * HD_ * HID_);
  f32_to_bf16_kernel<<<512, 256, 0, stream>>>(v_w, vwb, NKV_ * HD_ * HID_);
  f32_to_bf16_kernel<<<1024, 256, 0, stream>>>(o_w, owb, HID_ * NH_ * HD_);

  // 2) QKV projections (NT GEMMs), block tile 128x128
  gemm_nt_bf16_f32<<<dim3((NH_ * HD_) / 128, BS_ / 128), 128, 0, stream>>>(
      hsb, qwb, qf, BS_, NH_ * HD_, HID_);
  gemm_nt_bf16_f32<<<dim3((NKV_ * HD_) / 128, BS_ / 128), 128, 0, stream>>>(
      hsb, kwb, kf, BS_, NKV_ * HD_, HID_);
  gemm_nt_bf16_f32<<<dim3((NKV_ * HD_) / 128, BS_ / 128), 128, 0, stream>>>(
      hsb, vwb, vf, BS_, NKV_ * HD_, HID_);

  // 3) LayerNorm + RoPE, write bf16 Q/K ([b][h][s][d]) and transposed V ([b][kv][d][s])
  ln_rope_kernel<<<BS_, 128, 0, stream>>>(qf, kf, vf, cosb, sinb, q_nw, k_nw,
                                          qTb, kTb, vTb);

  // 4) causal flash attention -> ctx [b][s][h*d] bf16
  attn_kernel<<<(B_ * NH_ * (S_ / 16)) / 4, 128, 0, stream>>>(qTb, kTb, vTb, ctx);

  // 5) output projection -> fp32 d_out
  gemm_nt_bf16_f32<<<dim3(HID_ / 128, BS_ / 128), 128, 0, stream>>>(
      ctx, owb, out, BS_, HID_, NH_ * HD_);
}
